// InfoNCELoss_91293824843813
// MI455X (gfx1250) — matrix operand order repaired and checked by reading, not compiled, not run on hardware
//
#include <hip/hip_runtime.h>

typedef __attribute__((ext_vector_type(16))) _Float16 v16h;
typedef __attribute__((ext_vector_type(8)))  _Float16 v8h;
typedef __attribute__((ext_vector_type(8)))  float    v8f;

#define BZ 8192
#define DIM 256
#define INV_TEMP 100.0f      // 1 / 0.01
#define NCOLBLK 8            // column split: 8192 / 8 = 1024 cols per block
#define TILES_PER_BLK 64     // 1024 cols / 16
#define LDS_ROW 264          // 256 halves + 8 pad halves -> conflict-free ds_load_b128

// ---------------------------------------------------------------------------
// Kernel 1: row-wise L2 normalize f32 -> f16.  One wave per row.
// ---------------------------------------------------------------------------
__global__ __launch_bounds__(256) void nrm_kernel(const float* __restrict__ feat,
                                                  _Float16* __restrict__ outh) {
  const int row  = blockIdx.x * 8 + (threadIdx.x >> 5);
  const int lane = threadIdx.x & 31;
  const float4* src = (const float4*)(feat + (size_t)row * DIM + lane * 8);
  float4 v0 = src[0];
  float4 v1 = src[1];
  float ss = v0.x*v0.x + v0.y*v0.y + v0.z*v0.z + v0.w*v0.w
           + v1.x*v1.x + v1.y*v1.y + v1.z*v1.z + v1.w*v1.w;
  #pragma unroll
  for (int m = 16; m >= 1; m >>= 1) ss += __shfl_xor(ss, m, 32);
  const float inv = 1.0f / fmaxf(sqrtf(ss), 1e-12f);
  v8h o;
  o[0] = (_Float16)(v0.x * inv); o[1] = (_Float16)(v0.y * inv);
  o[2] = (_Float16)(v0.z * inv); o[3] = (_Float16)(v0.w * inv);
  o[4] = (_Float16)(v1.x * inv); o[5] = (_Float16)(v1.y * inv);
  o[6] = (_Float16)(v1.z * inv); o[7] = (_Float16)(v1.w * inv);
  *(v8h*)(outh + (size_t)row * DIM + lane * 8) = o;
}

// ---------------------------------------------------------------------------
// Kernel 2: f1 @ f2^T tile sweep with online logsumexp (logit domain).
//  grid = 64 rowBlks (128 rows) x 8 colBlks (1024 cols) = 512 blocks.
//  Per-wave state: mx (running max logit), sm (sumexp of 100*(x-mx)), dg
//  (diag logit, captured bitwise from the WMMA output -> acc = (dg == mx)).
// ---------------------------------------------------------------------------
__global__ __launch_bounds__(256) void logits_lse_kernel(const _Float16* __restrict__ f1h,
                                                         const _Float16* __restrict__ f2h,
                                                         float* __restrict__ partials) {
  __shared__ _Float16 bs[2][16 * LDS_ROW];   // 2 x 8448 B

  const int tid  = threadIdx.x;
  const int lane = tid & 31;
  const int wave = tid >> 5;
  const int lo   = lane & 15;
  const int hi   = lane >> 4;

  const int rowBlk  = blockIdx.x & 63;
  const int colBlk  = blockIdx.x >> 6;
  const int rowBase = rowBlk * 128 + wave * 16;   // this wave's 16 f1 rows
  const int arow    = rowBase + lo;
  const int tile0   = colBlk * TILES_PER_BLK;     // first 16-col tile index
  const int diagTile = rowBase >> 4;              // tile whose cols == our rows

  // Staging assignment: tile = 16 rows x 32 chunks of 16B; thread t handles
  // chunks t and t+256.
  const int c0row = tid >> 5,          c0off = (tid & 31) * 8;          // halves
  const int c1row = (tid + 256) >> 5,  c1off = ((tid + 256) & 31) * 8;

  // Preload the A tile (row arow, all K) in contiguous k order:
  // lane<16: K=[kb..kb+15]; lane>=16: K=[kb+16..kb+31].
  v16h a[8];
  #pragma unroll
  for (int k = 0; k < 8; ++k) {
    const int kb = 32 * k;
    a[k] = *(const v16h*)(f1h + (size_t)arow * DIM + kb + 16 * hi);
  }

  const float NEG_INF = -__builtin_huge_valf();
  float mx[8], sm[8], dg[8];               // per-lane rows m = r + 8*hi
  #pragma unroll
  for (int r = 0; r < 8; ++r) { mx[r] = NEG_INF; sm[r] = 0.0f; dg[r] = NEG_INF; }

  // Prologue: stage tile0 into buffer 0.
  {
    const size_t base = (size_t)(tile0 * 16) * DIM;
    uint4 g0 = *(const uint4*)(f2h + base + (size_t)c0row * DIM + c0off);
    uint4 g1 = *(const uint4*)(f2h + base + (size_t)c1row * DIM + c1off);
    *(uint4*)(&bs[0][c0row * LDS_ROW + c0off]) = g0;
    *(uint4*)(&bs[0][c1row * LDS_ROW + c1off]) = g1;
  }
  __syncthreads();

  for (int t = 0; t < TILES_PER_BLK; ++t) {
    // Issue staging loads for tile t+1 early (latency overlapped by compute).
    uint4 g0, g1;
    const bool more = (t + 1 < TILES_PER_BLK);
    if (more) {
      const size_t base = (size_t)((tile0 + t + 1) * 16) * DIM;
      g0 = *(const uint4*)(f2h + base + (size_t)c0row * DIM + c0off);
      g1 = *(const uint4*)(f2h + base + (size_t)c1row * DIM + c1off);
    }

    // Load the whole B tile for this wave first (one DS clause, one wait),
    // then run two independent WMMA accumulation chains.
    const _Float16* bp = &bs[t & 1][0];
    v16h b[8];
    #pragma unroll
    for (int k = 0; k < 8; ++k)
      __builtin_memcpy(&b[k], bp + lo * LDS_ROW + 32 * k + 16 * hi, 32);

    v8f c0v = {}, c1v = {};
    #pragma unroll
    for (int k = 0; k < 4; ++k) {
      c0v = __builtin_amdgcn_wmma_f32_16x16x32_f16(false, a[2*k],   false, b[2*k],
                                                   (short)0, c0v, false, false);
      c1v = __builtin_amdgcn_wmma_f32_16x16x32_f16(false, a[2*k+1], false, b[2*k+1],
                                                   (short)0, c1v, false, false);
    }
    v8f c = c0v + c1v;

    // Diagonal capture: wave-uniform — at most one tile per wave contains it.
    if (tile0 + t == diagTile) {
      #pragma unroll
      for (int r = 0; r < 8; ++r)
        if (lo == r + 8 * hi) dg[r] = c[r];   // bitwise element of this row
    }

    // Online softmax update in logit domain; exp arg scaled by 1/TEMP
    // (constant-folded into the exp2 multiplier).  One exp per element.
    #pragma unroll
    for (int r = 0; r < 8; ++r) {
      const float x  = c[r];
      const float mo = mx[r];
      const float m2 = fmaxf(mo, x);
      const float e  = __expf((fminf(mo, x) - m2) * INV_TEMP);
      const bool  up = (x > mo);
      sm[r] = __builtin_fmaf(sm[r], up ? e : 1.0f, up ? 1.0f : e);
      mx[r] = m2;
    }

    // Commit staged tile t+1 to the other buffer, then block barrier.
    if (more) {
      *(uint4*)(&bs[(t + 1) & 1][c0row * LDS_ROW + c0off]) = g0;
      *(uint4*)(&bs[(t + 1) & 1][c1row * LDS_ROW + c1off]) = g1;
    }
    __syncthreads();
  }

  // Combine across the 16 lanes sharing each row (xor 1,2,4,8 stays in half).
  #pragma unroll
  for (int msk = 1; msk <= 8; msk <<= 1) {
    #pragma unroll
    for (int r = 0; r < 8; ++r) {
      const float mo = __shfl_xor(mx[r], msk, 32);
      const float so = __shfl_xor(sm[r], msk, 32);
      const float m2 = fmaxf(mx[r], mo);
      sm[r] = sm[r] * __expf((mx[r] - m2) * INV_TEMP)
            + so    * __expf((mo    - m2) * INV_TEMP);
      mx[r] = m2;
      dg[r] = fmaxf(dg[r], __shfl_xor(dg[r], msk, 32));
    }
  }
  if (lo == 0) {    // lane 0 -> rows 0..7, lane 16 -> rows 8..15
    #pragma unroll
    for (int r = 0; r < 8; ++r) {
      const int grow = rowBase + r + 8 * hi;
      *(float4*)(partials + ((size_t)grow * NCOLBLK + colBlk) * 4) =
          make_float4(mx[r], sm[r], dg[r], 0.0f);
    }
  }
}

// ---------------------------------------------------------------------------
// Kernel 3: merge per-(row, colBlk) partials -> (loss, acc).  One block.
// ---------------------------------------------------------------------------
__global__ __launch_bounds__(1024) void combine_kernel(const float* __restrict__ partials,
                                                       float* __restrict__ out) {
  __shared__ float sl[32], sa[32];
  const float NEG_INF = -__builtin_huge_valf();
  float L = 0.0f, A = 0.0f;
  for (int row = threadIdx.x; row < BZ; row += 1024) {
    float M = NEG_INF, S = 0.0f, DG = NEG_INF;
    #pragma unroll
    for (int cB = 0; cB < NCOLBLK; ++cB) {
      const float4 st = *(const float4*)(partials + ((size_t)row * NCOLBLK + cB) * 4);
      const float m2 = fmaxf(M, st.x);
      S = S * __expf((M - m2) * INV_TEMP) + st.y * __expf((st.x - m2) * INV_TEMP);
      M = m2;
      DG = fmaxf(DG, st.z);
    }
    L += __logf(S) + (M - DG) * INV_TEMP;   // -log_softmax[positive]
    A += (DG == M) ? 1.0f : 0.0f;           // diag is argmax (ties -> index 0)
  }
  #pragma unroll
  for (int m = 16; m >= 1; m >>= 1) { L += __shfl_xor(L, m, 32); A += __shfl_xor(A, m, 32); }
  if ((threadIdx.x & 31) == 0) { sl[threadIdx.x >> 5] = L; sa[threadIdx.x >> 5] = A; }
  __syncthreads();
  if (threadIdx.x == 0) {
    float l = 0.0f, acc = 0.0f;
    for (int w = 0; w < 32; ++w) { l += sl[w]; acc += sa[w]; }
    out[0] = l / (float)BZ;
    out[1] = acc / (float)BZ;
  }
}

// ---------------------------------------------------------------------------
extern "C" void kernel_launch(void* const* d_in, const int* in_sizes, int n_in,
                              void* d_out, int out_size, void* d_ws, size_t ws_size,
                              hipStream_t stream) {
  const float* feature = (const float*)d_in[0];        // [2*BZ, DIM] f32
  float* out = (float*)d_out;                          // {loss, acc}

  _Float16* fh = (_Float16*)d_ws;                      // 2*BZ*DIM halves = 8 MB
  float* partials = (float*)((char*)d_ws +
                             (size_t)2 * BZ * DIM * sizeof(_Float16));  // 1 MB

  nrm_kernel<<<(2 * BZ) / 8, 256, 0, stream>>>(feature, fh);
  logits_lse_kernel<<<64 * NCOLBLK, 256, 0, stream>>>(fh, fh + (size_t)BZ * DIM, partials);
  combine_kernel<<<1, 1024, 0, stream>>>(partials, out);
}
